// EncoderBlock_77489799954613
// MI455X (gfx1250) — matrix operand order repaired and checked by reading, not compiled, not run on hardware
//
#include <hip/hip_runtime.h>

// ---------------------------------------------------------------------------
// MI455X (gfx1250) encoder block: bf16 WMMA GEMMs (async-LDS staged, double
// buffered, 64-wide K stages) + flash attention.
// B=4, S=2048, E=2048, H=16, HD=128, DH=256.
// ---------------------------------------------------------------------------

typedef __attribute__((ext_vector_type(16))) __bf16 v16bf;
typedef __attribute__((ext_vector_type(8)))  __bf16 v8bf;
typedef __attribute__((ext_vector_type(8)))  float  v8f;

#define WMMA_BF16(a, b, c) \
  __builtin_amdgcn_wmma_f32_16x16x32_bf16(false, (a), false, (b), (short)0, (c), false, false)

// A operand (16x32, M x K) from row-major [rows, ld]:
// lane l (0..15), half hf: row = row0+l, K chunks [k+hf*8 .. +7] and [k+16+hf*8 .. +7]
__device__ __forceinline__ v16bf load_fragA(const __bf16* base, int ld, int row, int k, int lane) {
  const int l = lane & 15, hf = (lane >> 4) & 1;
  const __bf16* p = base + (size_t)(row + l) * ld + k + hf * 8;
  v8bf lo = *(const v8bf*)p;
  v8bf hi = *(const v8bf*)(p + 16);
  return __builtin_shufflevector(lo, hi, 0,1,2,3,4,5,6,7,8,9,10,11,12,13,14,15);
}

// B operand (32x16, K x N) from row-major [N, K] source:
// lane l holds N = row0+l, K chunks [k+hf*8 .. +7] and [k+16+hf*8 .. +7].
// Two 16B loads so padded (non-32B) LDS row strides stay aligned.
__device__ __forceinline__ v16bf load_fragB(const __bf16* base, int ld, int row, int k, int lane) {
  const int l = lane & 15, hf = (lane >> 4) & 1;
  const __bf16* p = base + (size_t)(row + l) * ld + k + hf * 8;
  v8bf lo = *(const v8bf*)p;
  v8bf hi = *(const v8bf*)(p + 16);
  return __builtin_shufflevector(lo, hi, 0,1,2,3,4,5,6,7,8,9,10,11,12,13,14,15);
}

// ---------------------------------------------------------------------------
// fp32 -> bf16 conversion (weights)
// ---------------------------------------------------------------------------
__global__ void cvt_bf16_kernel(const float* __restrict__ in, __bf16* __restrict__ out, int n) {
  int i = blockIdx.x * 256 + threadIdx.x;
  if (i < n) out[i] = (__bf16)in[i];
}

// ---------------------------------------------------------------------------
// LayerNorm (fp32 in) -> bf16 out. One block (256 thr) per row of E=2048.
// ---------------------------------------------------------------------------
__global__ __launch_bounds__(256) void ln_bf16_kernel(
    const float* __restrict__ x, const float* __restrict__ w, const float* __restrict__ bta,
    __bf16* __restrict__ out) {
  const int E = 2048;
  const int row = blockIdx.x;
  const float* xr = x + (size_t)row * E;
  float sum = 0.f, sq = 0.f;
  for (int i = threadIdx.x; i < E; i += 256) { float v = xr[i]; sum += v; sq += v * v; }
  for (int off = 16; off >= 1; off >>= 1) {
    sum += __shfl_xor(sum, off, 32);
    sq  += __shfl_xor(sq,  off, 32);
  }
  __shared__ float ssum[8], ssq[8];
  const int wv = threadIdx.x >> 5, lane = threadIdx.x & 31;
  if (lane == 0) { ssum[wv] = sum; ssq[wv] = sq; }
  __syncthreads();
  sum = 0.f; sq = 0.f;
  #pragma unroll
  for (int i = 0; i < 8; ++i) { sum += ssum[i]; sq += ssq[i]; }
  const float mean = sum * (1.0f / E);
  const float var  = sq * (1.0f / E) - mean * mean;
  const float rstd = rsqrtf(var + 1e-5f);
  for (int i = threadIdx.x; i < E; i += 256)
    out[(size_t)row * E + i] = (__bf16)((xr[i] - mean) * rstd * w[i] + bta[i]);
}

// ---------------------------------------------------------------------------
// GEMM: C[M,N] = A[M,K](bf16) @ W[N,K]^T(bf16) + bias.
// mode 0: outB = bf16(C) (relu optional). mode 1: outF = C + resid (fp32).
// Block: 256 thr = 8 waves; block tile 256(M) x 64(N); wave tile 32x64.
// B panel staged in 64x64 K-stages via global_load_async_to_lds_b128 (2 per
// thread per stage), double buffered; steady-state loop peeled so the body is
// branch-free: issue-next / s_wait_asynccnt 2 / barrier / 16 WMMAs / barrier.
// LDS rows padded to 72 bf16 (144B) to spread banks.
// ---------------------------------------------------------------------------
#define BPAD 72

__global__ __launch_bounds__(256) void gemm_bf16_kernel(
    const __bf16* __restrict__ A, const __bf16* __restrict__ W,
    const float* __restrict__ bias, const float* __restrict__ resid,
    float* __restrict__ outF, __bf16* __restrict__ outB,
    int M, int N, int K, int mode, int relu) {
  const int wv = threadIdx.x >> 5, lane = threadIdx.x & 31;
  const int l = lane & 15, hf = (lane >> 4) & 1;
  const int m0 = blockIdx.y * 256 + wv * 32;
  const int n0 = blockIdx.x * 64;

  __shared__ __bf16 bsh[2][64 * BPAD];

  // async copy of one 64(N) x 64(K) B stage: each thread moves 32B (2 x b128)
  const int crow = threadIdx.x >> 2;            // 0..63 (N row within panel)
  const int ccol = (threadIdx.x & 3) * 16;      // 0,16,32,48 (bf16 elems)
  const unsigned long long wrow = (unsigned long long)(n0 + crow) * (unsigned)K;

  #define ISSUE_COPY(bufi, kk)                                                   \
    {                                                                            \
      unsigned long long g0 =                                                    \
          (unsigned long long)(size_t)(W + wrow + (kk) + ccol);                  \
      unsigned int l0 =                                                          \
          (unsigned int)(size_t)&bsh[bufi][crow * BPAD + ccol];                  \
      asm volatile("global_load_async_to_lds_b128 %0, %1, off"                   \
                   :: "v"(l0), "v"(g0) : "memory");                              \
      asm volatile("global_load_async_to_lds_b128 %0, %1, off"                   \
                   :: "v"(l0 + 16u), "v"(g0 + 16ull) : "memory");                \
    }

  v8f acc0[4] = {};
  v8f acc1[4] = {};

  auto compute_stage = [&](int buf, int kbase) {
    const __bf16* bp = &bsh[buf][0];
    #pragma unroll
    for (int ksub = 0; ksub < 2; ++ksub) {
      const int k = kbase + ksub * 32;
      v16bf a0 = load_fragA(A, K, m0, k, lane);
      v16bf a1 = load_fragA(A, K, m0 + 16, k, lane);
      #pragma unroll
      for (int t = 0; t < 4; ++t) {
        v16bf b = load_fragB(bp, BPAD, t * 16, ksub * 32, lane);
        acc0[t] = WMMA_BF16(a0, b, acc0[t]);
        acc1[t] = WMMA_BF16(a1, b, acc1[t]);
      }
    }
  };

  const int nstages = K >> 6;                   // 64-wide K stages
  ISSUE_COPY(0, 0)
  for (int ks = 0; ks < nstages - 1; ++ks) {
    ISSUE_COPY((ks + 1) & 1, (ks + 1) * 64)
    asm volatile("s_wait_asynccnt 0x2" ::: "memory");
    __syncthreads();
    compute_stage(ks & 1, ks * 64);
    __syncthreads();   // all waves done with this buffer before it refills
  }
  asm volatile("s_wait_asynccnt 0x0" ::: "memory");
  __syncthreads();
  compute_stage((nstages - 1) & 1, (nstages - 1) * 64);
  #undef ISSUE_COPY

  #pragma unroll
  for (int t = 0; t < 4; ++t) {
    const int col = n0 + t * 16 + l;
    const float bv = bias[col];
    #pragma unroll
    for (int half = 0; half < 2; ++half) {
      #pragma unroll
      for (int r = 0; r < 8; ++r) {
        const int row = m0 + half * 16 + hf * 8 + r;
        float v = (half ? acc1[t][r] : acc0[t][r]) + bv;
        const size_t idx = (size_t)row * N + col;
        if (mode == 0) {
          if (relu) v = fmaxf(v, 0.f);
          outB[idx] = (__bf16)v;
        } else {
          outF[idx] = v + resid[idx];
        }
      }
    }
  }
}

// ---------------------------------------------------------------------------
// Repack V section of qkv (bf16 [T,3E]) into Vt [B,H,HD,S] bf16
// ---------------------------------------------------------------------------
__global__ void repack_v_kernel(const __bf16* __restrict__ qkv, __bf16* __restrict__ vt) {
  const size_t total = (size_t)4 * 16 * 128 * 2048;
  size_t i = (size_t)blockIdx.x * 256 + threadIdx.x;
  if (i >= total) return;
  const int s = (int)(i & 2047); size_t j = i >> 11;
  const int d = (int)(j & 127);  j >>= 7;
  const int h = (int)(j & 15);   const int b = (int)(j >> 4);
  vt[i] = qkv[((size_t)(b * 2048 + s)) * 6144 + 4096 + h * 128 + d];
}

// ---------------------------------------------------------------------------
// Flash attention. One wave per (b, h, 16-query tile). 4 waves per block.
// scores = (Q K^T)/sqrt(128) + triu1(+1.0) + (-inf on padded keys); online
// softmax; P transposed C->A layout via LDS; ctx += P @ V using Vt.
// ---------------------------------------------------------------------------
__global__ __launch_bounds__(128) void flash_attn_kernel(
    const __bf16* __restrict__ qkv, const __bf16* __restrict__ vt,
    const int* __restrict__ amask, __bf16* __restrict__ ctx) {
  const int S = 2048, LDQ = 6144;
  const int wv = threadIdx.x >> 5, lane = threadIdx.x & 31;
  const int l = lane & 15, hf = (lane >> 4) & 1;

  const int tile = blockIdx.x * 4 + wv;     // 0 .. B*H*(S/16)-1 = 8191
  const int qt = tile & 127;
  const int bh = tile >> 7;
  const int h = bh & 15, b = bh >> 4;
  const int q0 = qt * 16;

  const __bf16* qbase = qkv + (size_t)b * S * LDQ + h * 128;
  const __bf16* kbase = qkv + (size_t)b * S * LDQ + 2048 + h * 128;
  const __bf16* vbase = vt + ((size_t)(b * 16 + h)) * 128 * S;   // [128][S]
  const int* mrow_mask = amask + b * S;

  v16bf aQ[4];
  #pragma unroll
  for (int d = 0; d < 4; ++d) aQ[d] = load_fragA(qbase, LDQ, q0, d * 32, lane);

  v8f acc[8] = {};
  float mmax[8], lsum[8];
  #pragma unroll
  for (int r = 0; r < 8; ++r) { mmax[r] = -3.0e38f; lsum[r] = 0.f; }

  __shared__ __bf16 psh[4][16 * 32];
  __bf16* pp = &psh[wv][0];

  const float scale = 0.08838834764831845f;   // 1/sqrt(128)

  for (int c0 = 0; c0 < S; c0 += 32) {
    v8f sc[2];
    #pragma unroll
    for (int sub = 0; sub < 2; ++sub) {
      const int n0 = c0 + sub * 16;
      v8f s = {};
      #pragma unroll
      for (int dk = 0; dk < 4; ++dk) {
        v16bf bK = load_fragB(kbase, LDQ, n0, dk * 32, lane);
        s = WMMA_BF16(aQ[dk], bK, s);
      }
      const int key = n0 + l;
      const float madd = (mrow_mask[key] == 0) ? -1.0e30f : 0.f;
      #pragma unroll
      for (int r = 0; r < 8; ++r) {
        const int qrow = q0 + hf * 8 + r;
        s[r] = s[r] * scale + ((key > qrow) ? 1.0f : 0.f) + madd;
      }
      sc[sub] = s;
    }

    float cm[8];
    #pragma unroll
    for (int r = 0; r < 8; ++r) cm[r] = fmaxf(sc[0][r], sc[1][r]);
    #pragma unroll
    for (int off = 8; off >= 1; off >>= 1)
      #pragma unroll
      for (int r = 0; r < 8; ++r) cm[r] = fmaxf(cm[r], __shfl_xor(cm[r], off, 32));

    float fac[8];
    #pragma unroll
    for (int r = 0; r < 8; ++r) {
      const float nm = fmaxf(mmax[r], cm[r]);
      fac[r] = __expf(mmax[r] - nm);
      mmax[r] = nm;
      lsum[r] *= fac[r];
    }
    #pragma unroll
    for (int t = 0; t < 8; ++t)
      #pragma unroll
      for (int r = 0; r < 8; ++r) acc[t][r] *= fac[r];

    float rs[8];
    #pragma unroll
    for (int r = 0; r < 8; ++r) rs[r] = 0.f;
    #pragma unroll
    for (int sub = 0; sub < 2; ++sub)
      #pragma unroll
      for (int r = 0; r < 8; ++r) {
        const float p = __expf(sc[sub][r] - mmax[r]);
        rs[r] += p;
        pp[(hf * 8 + r) * 32 + sub * 16 + l] = (__bf16)p;
      }
    #pragma unroll
    for (int off = 8; off >= 1; off >>= 1)
      #pragma unroll
      for (int r = 0; r < 8; ++r) rs[r] += __shfl_xor(rs[r], off, 32);
    #pragma unroll
    for (int r = 0; r < 8; ++r) lsum[r] += rs[r];

    // intra-wave LDS RAW: DS ops are in-order per wave; wait stores, then read
    asm volatile("s_wait_dscnt 0x0" ::: "memory");
    v16bf pa = load_fragA(pp, 32, 0, 0, lane);

    #pragma unroll
    for (int dt = 0; dt < 8; ++dt) {
      v16bf bV = load_fragB(vbase, S, dt * 16, c0, lane);
      acc[dt] = WMMA_BF16(pa, bV, acc[dt]);
    }
  }

  #pragma unroll
  for (int r = 0; r < 8; ++r) {
    const float inv = 1.0f / lsum[r];
    const size_t row = (size_t)(b * S + q0 + hf * 8 + r);
    #pragma unroll
    for (int dt = 0; dt < 8; ++dt)
      ctx[row * 2048 + h * 128 + dt * 16 + l] = (__bf16)(acc[dt][r] * inv);
  }
}

// ---------------------------------------------------------------------------
// Host launcher
// ---------------------------------------------------------------------------
extern "C" void kernel_launch(void* const* d_in, const int* in_sizes, int n_in,
                              void* d_out, int out_size, void* d_ws, size_t ws_size,
                              hipStream_t stream) {
  (void)in_sizes; (void)n_in; (void)out_size; (void)ws_size;
  const int Bb = 4, S = 2048, E = 2048, H = 16, HD = 128, DH = 256;
  const size_t T = (size_t)Bb * S;   // 8192 tokens

  const float* x        = (const float*)d_in[0];
  const int*   amask    = (const int*)  d_in[1];
  const float* ln1_w    = (const float*)d_in[2];
  const float* ln1_b    = (const float*)d_in[3];
  const float* in_w     = (const float*)d_in[4];
  const float* in_b     = (const float*)d_in[5];
  const float* out_w    = (const float*)d_in[6];
  const float* out_b    = (const float*)d_in[7];
  const float* ln2_w    = (const float*)d_in[8];
  const float* ln2_b    = (const float*)d_in[9];
  const float* w1       = (const float*)d_in[10];
  const float* b1       = (const float*)d_in[11];
  const float* w2       = (const float*)d_in[12];
  const float* b2       = (const float*)d_in[13];
  float* out = (float*)d_out;

  char* ws = (char*)d_ws;
  size_t off = 0;
  auto carve = [&](size_t bytes) { char* p = ws + off; off = (off + bytes + 255) & ~(size_t)255; return p; };
  __bf16* hb    = (__bf16*)carve(T * E * 2);
  __bf16* wqkvb = (__bf16*)carve((size_t)3 * E * E * 2);
  __bf16* woutb = (__bf16*)carve((size_t)E * E * 2);
  __bf16* w1b   = (__bf16*)carve((size_t)DH * E * 2);
  __bf16* w2b   = (__bf16*)carve((size_t)E * DH * 2);
  __bf16* qkvb  = (__bf16*)carve(T * 3 * E * 2);
  __bf16* vtb   = (__bf16*)carve((size_t)Bb * H * HD * S * 2);
  __bf16* ctxb  = (__bf16*)carve(T * E * 2);
  float*  x2    = (float*)carve(T * E * 4);
  __bf16* a1b   = (__bf16*)carve(T * DH * 2);

  cvt_bf16_kernel<<<(3 * E * E + 255) / 256, 256, 0, stream>>>(in_w,  wqkvb, 3 * E * E);
  cvt_bf16_kernel<<<(E * E + 255) / 256,     256, 0, stream>>>(out_w, woutb, E * E);
  cvt_bf16_kernel<<<(DH * E + 255) / 256,    256, 0, stream>>>(w1,    w1b,   DH * E);
  cvt_bf16_kernel<<<(E * DH + 255) / 256,    256, 0, stream>>>(w2,    w2b,   E * DH);

  ln_bf16_kernel<<<(int)T, 256, 0, stream>>>(x, ln1_w, ln1_b, hb);

  {  // QKV GEMM
    dim3 g(3 * E / 64, (int)T / 256);
    gemm_bf16_kernel<<<g, 256, 0, stream>>>(hb, wqkvb, in_b, nullptr, nullptr, qkvb,
                                            (int)T, 3 * E, E, 0, 0);
  }

  repack_v_kernel<<<(int)(((size_t)Bb * H * HD * S + 255) / 256), 256, 0, stream>>>(qkvb, vtb);

  flash_attn_kernel<<<Bb * H * (S / 16) / 4, 128, 0, stream>>>(qkvb, vtb, amask, ctxb);

  {  // out-proj + residual(x)
    dim3 g(E / 64, (int)T / 256);
    gemm_bf16_kernel<<<g, 256, 0, stream>>>(ctxb, woutb, out_b, x, x2, nullptr,
                                            (int)T, E, E, 1, 0);
  }

  ln_bf16_kernel<<<(int)T, 256, 0, stream>>>(x2, ln2_w, ln2_b, hb);

  {  // MLP fc1 + ReLU
    dim3 g(DH / 64, (int)T / 256);
    gemm_bf16_kernel<<<g, 256, 0, stream>>>(hb, w1b, b1, nullptr, nullptr, a1b,
                                            (int)T, DH, E, 0, 1);
  }

  {  // MLP fc2 + residual(x2)
    dim3 g(E / 64, (int)T / 256);
    gemm_bf16_kernel<<<g, 256, 0, stream>>>(a1b, w2b, b2, x2, out, nullptr,
                                            (int)T, E, DH, 1, 0);
  }
}
